// ConvCorrectModule_72172630442485
// MI455X (gfx1250) — matrix operand order repaired and checked by reading, not compiled, not run on hardware
//
#include <hip/hip_runtime.h>
#include <hip/hip_bf16.h>

// ---------------- constants (match setup_inputs) ----------------
#define NHEADS 6
#define NPOINTS 4
static constexpr int Bc   = 8;     // batch
static constexpr int Hc   = 32;    // query H
static constexpr int Wc   = 32;    // query W
static constexpr int LQ   = 3 * Hc * Wc;   // 3072
static constexpr int Cc   = 768;
static constexpr int DHEAD = Cc / NHEADS;  // 128
static constexpr int LIN  = 64 * 64;       // 4096 (feat spatial)
static constexpr int HID  = Cc / 4;        // 192

typedef __attribute__((ext_vector_type(16))) __bf16 v16bf;
typedef __attribute__((ext_vector_type(8)))  __bf16 v8bf;
typedef __attribute__((ext_vector_type(8)))  float  v8f;
typedef __attribute__((ext_vector_type(4)))  float  f4v;

// ---------------- LayerNorm: one block (256 threads) per row ----------------
__global__ __launch_bounds__(256)
void ln_kernel(const float* __restrict__ x, const float* __restrict__ w,
               const float* __restrict__ b, float* __restrict__ y, int C) {
  const int row = blockIdx.x;
  const float* xr = x + (size_t)row * C;
  float*       yr = y + (size_t)row * C;
  __shared__ float red[256];
  float s = 0.f;
  for (int c = threadIdx.x; c < C; c += 256) s += xr[c];
  red[threadIdx.x] = s; __syncthreads();
  for (int st = 128; st > 0; st >>= 1) {
    if (threadIdx.x < st) red[threadIdx.x] += red[threadIdx.x + st];
    __syncthreads();
  }
  const float mean = red[0] / (float)C;
  __syncthreads();
  float v = 0.f;
  for (int c = threadIdx.x; c < C; c += 256) { float t = xr[c] - mean; v += t * t; }
  red[threadIdx.x] = v; __syncthreads();
  for (int st = 128; st > 0; st >>= 1) {
    if (threadIdx.x < st) red[threadIdx.x] += red[threadIdx.x + st];
    __syncthreads();
  }
  const float rstd = rsqrtf(red[0] / (float)C + 1e-6f);
  for (int c = threadIdx.x; c < C; c += 256)
    yr[c] = (xr[c] - mean) * rstd * w[c] + b[c];
}

// ---------------- BF16 WMMA GEMM: out = A(MxK) @ B(KxN) + bias (+res) -------
// REQUIRES (guaranteed by all call sites): M % 128 == 0, K % 32 == 0,
// bias != nullptr.  N may be ragged (48 / 24 / 192) -> handled branchlessly.
// Tile: 128 (M) x 128 (N), K stepped by 32; 256 threads = 8 waves (4x2 grid,
// each wave owns 2x4 16x16 fragments -> 8 WMMA per K-step). Double-buffered
// LDS tiles stored directly in WMMA fragment layout (ISA 7.12.2):
//   A 16x32 bf16 frag: lane = m + 16*khalf, elem e = grp*8 + (rem&7)
//   B 32x16 bf16 frag: lane = n + 16*((k>>3)&1), elem e = (k>>4)*8 + (k&7)
// so every bf16 run a thread produces is a contiguous 16B/32B LDS store and
// each lane's 16-bf16 fragment is one contiguous 32B aligned LDS read.
__global__ __launch_bounds__(256)
void gemm_wmma_bf16(const float* __restrict__ A, const float* __restrict__ Bw,
                    const float* __restrict__ bias, const float* __restrict__ res,
                    float* __restrict__ out, int M, int N, int K) {
  __shared__ __align__(32) __bf16 Al[2][8][32][16];  // 16 KB (double buffered)
  __shared__ __align__(32) __bf16 Bl[2][8][32][16];  // 16 KB
  const int tid  = threadIdx.x;
  const int lane = tid & 31;
  const int wave = tid >> 5;
  const int wm = wave >> 1;   // 0..3 -> mfrag pair (rows wm*32..wm*32+31)
  const int wn = wave & 1;    // 0..1 -> nfrag quad (cols wn*64..wn*64+63)
  const int m0 = blockIdx.y * 128;
  const int n0 = blockIdx.x * 128;

  // A loader: thread -> (row = m0 + tid>>1, k-half = tid&1): 16 floats along K.
  const int ar = tid >> 1;
  const int ah = tid & 1;
  const int atm = ar >> 4, am = ar & 15;
  // B loader: thread -> (n = tid&127, k-half = tid>>7): 16 floats along K.
  const int bn  = tid & 127;
  const int bkq = tid >> 7;                  // 0..1
  const int btn = bn >> 4;
  const int bLane = (bn & 15) + (bkq << 4);

  f4v   aReg[4];
  float bReg[16];

  const int bcol = n0 + bn;
  const int bcc  = bcol < N ? bcol : N - 1;  // clamp; value zeroed below
  const float* aBase = A + (size_t)(m0 + ar) * K + ah * 16;

  auto fetchTile = [&](int k0) {
    const float* ap = aBase + k0;
    #pragma unroll
    for (int j = 0; j < 4; ++j) aReg[j] = *(const f4v*)(ap + 4 * j);
    const float* bp = Bw + (size_t)(k0 + bkq * 8) * N + bcc;
    #pragma unroll
    for (int j = 0; j < 16; ++j) {
      float v = bp[(size_t)(j + (j & 8)) * N];            // unconditional (clamped addr)
      bReg[j] = (bcol < N) ? v : 0.f;         // branchless zero for OOB col
    }
  };
  auto stageTile = [&](int buf) {
    float f[16];
    #pragma unroll
    for (int j = 0; j < 4; ++j) {
      f[4 * j + 0] = aReg[j][0]; f[4 * j + 1] = aReg[j][1];
      f[4 * j + 2] = aReg[j][2]; f[4 * j + 3] = aReg[j][3];
    }
    v8bf lo, hi;
    v16bf bpk;
    #pragma unroll
    for (int j = 0; j < 8; ++j) {
      lo[j] = (__bf16)f[j];        // rem 0..7  -> lane am,    e = ah*8 + j
      hi[j] = (__bf16)f[8 + j];    // rem 8..15 -> lane am+16, e = ah*8 + j
    }
    #pragma unroll
    for (int j = 0; j < 16; ++j) bpk[j] = (__bf16)bReg[j];  // e = j
    *(v8bf*)&Al[buf][atm][am][ah * 8]      = lo;
    *(v8bf*)&Al[buf][atm][am + 16][ah * 8] = hi;
    *(v16bf*)&Bl[buf][btn][bLane][0]       = bpk;
  };

  const int nsteps = K >> 5;  // K % 32 == 0
  fetchTile(0);
  stageTile(0);
  __syncthreads();

  v8f acc[2][4] = {};
  int p = 0;
  for (int step = 0; step < nsteps; ++step) {
    const bool more = (step + 1) < nsteps;
    if (more) fetchTile((step + 1) << 5);          // overlap VMEM with WMMA
    if (step + 2 < nsteps)
      __builtin_prefetch(aBase + ((step + 2) << 5), 0, 1);

    v16bf af0 = *(const v16bf*)&Al[p][wm * 2 + 0][lane][0];
    v16bf af1 = *(const v16bf*)&Al[p][wm * 2 + 1][lane][0];
    v16bf bf[4];
    #pragma unroll
    for (int q = 0; q < 4; ++q)
      bf[q] = *(const v16bf*)&Bl[p][wn * 4 + q][lane][0];

    #pragma unroll
    for (int q = 0; q < 4; ++q) {
      acc[0][q] = __builtin_amdgcn_wmma_f32_16x16x32_bf16(
          false, af0, false, bf[q], (short)0, acc[0][q], false, false);
      acc[1][q] = __builtin_amdgcn_wmma_f32_16x16x32_bf16(
          false, af1, false, bf[q], (short)0, acc[1][q], false, false);
    }

    if (more) stageTile(p ^ 1);   // write other buffer: no extra barrier needed
    __syncthreads();
    p ^= 1;
  }

  // ---- epilogue: f32 C/D layout (lane -> N, vgpr r -> M); rows always valid.
  // res / no-res variants hoisted so loads stay unconditional and pipelined.
  const int colBase = n0 + wn * 64 + (lane & 15);
  const int rowBase = m0 + wm * 32 + ((lane >> 4) << 3);
  if (res) {
    #pragma unroll
    for (int a = 0; a < 2; ++a) {
      #pragma unroll
      for (int b2 = 0; b2 < 4; ++b2) {
        const int col = colBase + b2 * 16;
        if (col >= N) continue;
        const float bv = bias[col];
        const size_t base = (size_t)(rowBase + a * 16) * N + col;
        float rv[8];
        #pragma unroll
        for (int r = 0; r < 8; ++r) rv[r] = res[base + (size_t)r * N];
        #pragma unroll
        for (int r = 0; r < 8; ++r)
          out[base + (size_t)r * N] = acc[a][b2][r] + bv + rv[r];
      }
    }
  } else {
    #pragma unroll
    for (int a = 0; a < 2; ++a) {
      #pragma unroll
      for (int b2 = 0; b2 < 4; ++b2) {
        const int col = colBase + b2 * 16;
        if (col >= N) continue;
        const float bv = bias[col];
        const size_t base = (size_t)(rowBase + a * 16) * N + col;
        #pragma unroll
        for (int r = 0; r < 8; ++r)
          out[base + (size_t)r * N] = acc[a][b2][r] + bv;
      }
    }
  }
}

// ---------------- sampling prep: softmax(aw) and loc = ref + off/norm -------
__global__ __launch_bounds__(256)
void prep_sampling(const float* __restrict__ so_out, const float* __restrict__ aw_out,
                   const float* __restrict__ refp, const int* __restrict__ sshapes,
                   float* __restrict__ loc, float* __restrict__ aww, int total) {
  const int t = blockIdx.x * 256 + threadIdx.x;
  if (t >= total) return;                 // total = B*Lq*nH
  const int h  = t % NHEADS;
  const int bq = t / NHEADS;
  const int Hl = sshapes[0], Wl = sshapes[1];
  const float rx = refp[(size_t)bq * 2 + 0];
  const float ry = refp[(size_t)bq * 2 + 1];
  const float* so = so_out + (size_t)bq * (NHEADS * NPOINTS * 2) + h * (NPOINTS * 2);
  const float* aw = aw_out + (size_t)bq * (NHEADS * NPOINTS) + h * NPOINTS;
  float a[NPOINTS], mx = -1e30f;
  #pragma unroll
  for (int p = 0; p < NPOINTS; ++p) { a[p] = aw[p]; mx = fmaxf(mx, a[p]); }
  float ssum = 0.f;
  #pragma unroll
  for (int p = 0; p < NPOINTS; ++p) { a[p] = expf(a[p] - mx); ssum += a[p]; }
  const float inv = 1.f / ssum;
  float* lo = loc + (size_t)t * (NPOINTS * 2);
  float* ao = aww + (size_t)t * NPOINTS;
  #pragma unroll
  for (int p = 0; p < NPOINTS; ++p) {
    lo[p * 2 + 0] = rx + so[p * 2 + 0] / (float)Wl;
    lo[p * 2 + 1] = ry + so[p * 2 + 1] / (float)Hl;
    ao[p] = a[p] * inv;
  }
}

// ---------------- bilinear gather + attention-weight sum --------------------
// block = one (b,q,h); threads = DHEAD. Coalesced along head dim.
__global__ __launch_bounds__(DHEAD)
void attn_gather(const float* __restrict__ value, const float* __restrict__ loc,
                 const float* __restrict__ aww, const int* __restrict__ sshapes,
                 float* __restrict__ attn_in) {
  const int blk = blockIdx.x;            // (b*Lq+q)*nH + h
  const int dd  = threadIdx.x;
  const int h   = blk % NHEADS;
  const int bq  = blk / NHEADS;
  const int b   = bq / LQ;
  __shared__ float sl[NPOINTS * 2];
  __shared__ float sa[NPOINTS];
  if (dd < NPOINTS * 2) sl[dd] = loc[(size_t)blk * (NPOINTS * 2) + dd];
  if (dd < NPOINTS)     sa[dd] = aww[(size_t)blk * NPOINTS + dd];
  __syncthreads();
  const int Hl = sshapes[0], Wl = sshapes[1];
  const float* vb = value + (size_t)b * Hl * Wl * Cc + h * DHEAD + dd;
  float acc = 0.f;
  #pragma unroll
  for (int p = 0; p < NPOINTS; ++p) {
    const float x = sl[p * 2 + 0] * (float)Wl - 0.5f;
    const float y = sl[p * 2 + 1] * (float)Hl - 0.5f;
    const float x0f = floorf(x), y0f = floorf(y);
    const float wx1 = x - x0f, wy1 = y - y0f;
    const int xi0 = (int)x0f, yi0 = (int)y0f;
    const float ap = sa[p];
    #pragma unroll
    for (int dy = 0; dy < 2; ++dy) {
      #pragma unroll
      for (int dx = 0; dx < 2; ++dx) {
        const int xi = xi0 + dx, yi = yi0 + dy;
        const float w = (dy ? wy1 : 1.f - wy1) * (dx ? wx1 : 1.f - wx1);
        if (xi >= 0 && xi < Wl && yi >= 0 && yi < Hl)
          acc += ap * w * vb[(size_t)(yi * Wl + xi) * Cc];
      }
    }
  }
  attn_in[(size_t)bq * Cc + h * DHEAD + dd] = acc;
}

// ---------------- depthwise 3x3 conv (SAME, zero pad) + exact GELU ----------
// x layout: (img = b*3+c, pix = y*W+x, ch) row-major, same layout for y.
__global__ __launch_bounds__(HID)
void dwconv_gelu(const float* __restrict__ x, const float* __restrict__ w,
                 const float* __restrict__ bias, float* __restrict__ y) {
  const int ch  = threadIdx.x;
  const int pix = blockIdx.x % (Hc * Wc);
  const int img = blockIdx.x / (Hc * Wc);
  const int px = pix % Wc, py = pix / Wc;
  const float* xi = x + (size_t)img * (Hc * Wc) * HID;
  float acc = bias[ch];
  #pragma unroll
  for (int ky = -1; ky <= 1; ++ky) {
    const int yy = py + ky;
    if (yy < 0 || yy >= Hc) continue;
    #pragma unroll
    for (int kx = -1; kx <= 1; ++kx) {
      const int xx = px + kx;
      if (xx < 0 || xx >= Wc) continue;
      acc += w[ch * 9 + (ky + 1) * 3 + (kx + 1)] * xi[(size_t)(yy * Wc + xx) * HID + ch];
    }
  }
  const float g = 0.5f * acc * (1.f + erff(acc * 0.70710678118654752440f));
  y[(size_t)img * (Hc * Wc) * HID + (size_t)pix * HID + ch] = g;
}

// ---------------- host orchestration ----------------------------------------
extern "C" void kernel_launch(void* const* d_in, const int* in_sizes, int n_in,
                              void* d_out, int out_size, void* d_ws, size_t ws_size,
                              hipStream_t stream) {
  (void)in_sizes; (void)n_in; (void)out_size; (void)ws_size;
  const float* query = (const float*)d_in[0];
  const float* refp  = (const float*)d_in[1];
  const float* feat  = (const float*)d_in[2];
  const int*   sshp  = (const int*)  d_in[3];
  const float* qn_w  = (const float*)d_in[7];
  const float* qn_b  = (const float*)d_in[8];
  const float* fn_w  = (const float*)d_in[9];
  const float* fn_b  = (const float*)d_in[10];
  const float* so_w  = (const float*)d_in[11];
  const float* so_b  = (const float*)d_in[12];
  const float* aw_w  = (const float*)d_in[13];
  const float* aw_b  = (const float*)d_in[14];
  const float* vp_w  = (const float*)d_in[15];
  const float* vp_b  = (const float*)d_in[16];
  const float* op_w  = (const float*)d_in[17];
  const float* op_b  = (const float*)d_in[18];
  const float* ffn_w = (const float*)d_in[19];
  const float* ffn_b = (const float*)d_in[20];
  const float* fc1_w = (const float*)d_in[21];
  const float* fc1_b = (const float*)d_in[22];
  const float* dw_w  = (const float*)d_in[23];
  const float* dw_b  = (const float*)d_in[24];
  const float* fc2_w = (const float*)d_in[25];
  const float* fc2_b = (const float*)d_in[26];
  float* out = (float*)d_out;

  const int BLq = Bc * LQ;      // 24576 query rows
  const int BLi = Bc * LIN;     // 32768 feat rows

  // workspace layout (float units)
  float* ws = (float*)d_ws;
  const size_t SZ_QLN = (size_t)BLq * Cc;   // 18,874,368
  const size_t SZ_FLN = (size_t)BLi * Cc;   // 25,165,824
  float* qln     = ws;                       // later reused as ffn_ln
  float* fln     = qln + SZ_QLN;             // later reused as attn_in
  float* value   = fln + SZ_FLN;             // later reused for fc1/gelu
  float* so_out  = value + SZ_FLN;           // (BLq, 48)
  float* aw_out  = so_out + (size_t)BLq * (NHEADS * NPOINTS * 2);
  float* loc     = aw_out + (size_t)BLq * (NHEADS * NPOINTS);
  float* aww     = loc + (size_t)BLq * NHEADS * NPOINTS * 2;
  float* attn_in = fln;                      // reuse (f_ln dead after value GEMM)
  float* ffn_ln  = qln;                      // reuse (q_ln dead after offs/aw GEMMs)
  float* fc1_out = value;                    // reuse (value dead after gather)
  float* dwg_out = value + (size_t)BLq * HID;

  // 1) LayerNorms
  ln_kernel<<<BLq, 256, 0, stream>>>(query, qn_w, qn_b, qln, Cc);
  ln_kernel<<<BLi, 256, 0, stream>>>(feat,  fn_w, fn_b, fln, Cc);

  // 2) value = f_ln @ vp_w + vp_b           (32768 x 768 x 768)
  gemm_wmma_bf16<<<dim3(Cc / 128, BLi / 128), 256, 0, stream>>>(
      fln, vp_w, vp_b, nullptr, value, BLi, Cc, Cc);

  // 3) sampling offsets / attention logits  (24576 x {48,24} x 768)
  gemm_wmma_bf16<<<dim3(1, BLq / 128), 256, 0, stream>>>(
      qln, so_w, so_b, nullptr, so_out, BLq, NHEADS * NPOINTS * 2, Cc);
  gemm_wmma_bf16<<<dim3(1, BLq / 128), 256, 0, stream>>>(
      qln, aw_w, aw_b, nullptr, aw_out, BLq, NHEADS * NPOINTS, Cc);

  // 4) softmax + sampling locations
  {
    const int total = BLq * NHEADS;
    prep_sampling<<<(total + 255) / 256, 256, 0, stream>>>(
        so_out, aw_out, refp, sshp, loc, aww, total);
  }

  // 5) bilinear gather + weight-sum -> attn_in (B,Lq,C)
  attn_gather<<<BLq * NHEADS, DHEAD, 0, stream>>>(value, loc, aww, sshp, attn_in);

  // 6) d_out = attn_in @ op_w + op_b + query   (residual; fully overwrites d_out)
  gemm_wmma_bf16<<<dim3(Cc / 128, BLq / 128), 256, 0, stream>>>(
      attn_in, op_w, op_b, query, out, BLq, Cc, Cc);

  // 7) ffn_ln = LN(d_out)
  ln_kernel<<<BLq, 256, 0, stream>>>(out, ffn_w, ffn_b, ffn_ln, Cc);

  // 8) fc1: (24576 x 192 x 768), N ragged over 2 block-cols
  gemm_wmma_bf16<<<dim3((HID + 127) / 128, BLq / 128), 256, 0, stream>>>(
      ffn_ln, fc1_w, fc1_b, nullptr, fc1_out, BLq, HID, Cc);

  // 9) depthwise 3x3 + GELU
  dwconv_gelu<<<Bc * 3 * Hc * Wc, HID, 0, stream>>>(fc1_out, dw_w, dw_b, dwg_out);

  // 10) d_out += gelu @ fc2_w + fc2_b        (24576 x 768 x 192, residual from d_out)
  gemm_wmma_bf16<<<dim3(Cc / 128, BLq / 128), 256, 0, stream>>>(
      dwg_out, fc2_w, fc2_b, out, out, BLq, Cc, HID);
}
